// WindowedMultiHeadSelfAttention_83562883711458
// MI455X (gfx1250) — matrix operand order repaired and checked by reading, not compiled, not run on hardware
//
#include <hip/hip_runtime.h>
#include <hip/hip_bf16.h>

typedef __attribute__((ext_vector_type(16))) __bf16 v16bf;
typedef __attribute__((ext_vector_type(8)))  __bf16 v8bf;
typedef __attribute__((ext_vector_type(8)))  float  v8f;
typedef __attribute__((ext_vector_type(4)))  float  v4f;

#define NW 8   // waves per 256-thread workgroup (wave32)

// ---------------- WMMA tile loaders (CDNA5 bf16 16x16x32 layouts) -------------
// A-matrix 16x32 (MxK): lane<16 -> halves0-7:K=0..7, halves8-15:K=16..23
//                       lane>=16 -> K=8..15 / 24..31 ; row M = lane&15
static __device__ __forceinline__ v16bf load_a16x32(const __bf16* base, int ld,
                                                    int m0, int k0, int lane) {
  int m  = lane & 15;
  int kh = (lane & 16) >> 1;            // 0 or 8
  const __bf16* p = base + (m0 + m) * ld + k0 + kh;
  v8bf lo = *(const v8bf*)(p);
  v8bf hi = *(const v8bf*)(p + 16);
  v16bf r;
#pragma unroll
  for (int i = 0; i < 8; ++i) { r[i] = lo[i]; r[i + 8] = hi[i]; }
  return r;
}

// B-matrix 32x16 (KxN): lane = column N (mod 16); lanes<16 hold K=0..7/16..23,
// lanes>=16 hold K=8..15/24..31.  Source laid out BT[n][k] (k contiguous).
static __device__ __forceinline__ v16bf load_b32x16(const __bf16* baseT, int ld,
                                                    int n0, int k0, int lane) {
  int n  = lane & 15;
  int kh = (lane & 16) >> 1;            // 0 or 8
  const __bf16* p = baseT + (n0 + n) * ld + k0 + kh;
  v8bf lo = *(const v8bf*)(p);
  v8bf hi = *(const v8bf*)(p + 16);
  v16bf r;
#pragma unroll
  for (int i = 0; i < 8; ++i) { r[i] = lo[i]; r[i + 8] = hi[i]; }
  return r;
}

static __device__ __forceinline__ v8f wmma_bf16(v16bf a, v16bf b, v8f c) {
  return __builtin_amdgcn_wmma_f32_16x16x32_bf16(false, a, false, b,
                                                 (short)0, c, false, false);
}

// ---------------- weight convert + transpose: f32 [k][n] -> bf16 [n][k] ------
__global__ __launch_bounds__(256) void convert_weights(
    const float* __restrict__ Wqkv, const float* __restrict__ Wproj,
    __bf16* __restrict__ WqkvT, __bf16* __restrict__ WprojT) {
  int gid = blockIdx.x * blockDim.x + threadIdx.x;
  const int NQ = 512 * 1536;
  if (gid < NQ) {
    int k = gid / 1536, n = gid % 1536;
    WqkvT[n * 512 + k] = (__bf16)Wqkv[gid];
  } else {
    int g = gid - NQ;
    if (g < 512 * 512) {
      int k = g >> 9, n = g & 511;
      WprojT[n * 512 + k] = (__bf16)Wproj[g];
    }
  }
}

// ---------------- fused windowed MHSA: one workgroup = one 8x8 window --------
__global__ __launch_bounds__(256) void wmsa_kernel(
    const float* __restrict__ x,
    const __bf16* __restrict__ WqkvT,   // [1536][512] bf16
    const __bf16* __restrict__ WprojT,  // [512][512]  bf16
    const float* __restrict__ b_qkv,
    const float* __restrict__ b_proj,
    float* __restrict__ out) {
  extern __shared__ char smem_raw[];
  __bf16* Xb = (__bf16*)smem_raw;                // 64x512 bf16 (64KB); reused as
  float*  Sc = (float*)smem_raw;                 //   4 x 64x64 f32 scores (64KB)
  __bf16* Qb = (__bf16*)(smem_raw + 65536);      // 64x512 Q (pre-scaled); later O
  __bf16* Kb = (__bf16*)(smem_raw + 131072);     // 64x512 K
  __bf16* Vt = (__bf16*)(smem_raw + 196608);     // 8 heads x 64(d) x 64(t)

  const int w    = blockIdx.x;                   // window id 0..2047
  const int b    = w >> 8;
  const int wh   = (w >> 4) & 15;
  const int ww   = w & 15;
  const int tid  = threadIdx.x;
  const int wv   = tid >> 5;
  const int lane = tid & 31;
  const long winBase = (((long)b * 128 + wh * 8) * 128 + ww * 8) * 512;

  // ---- Stage A: load 64x512 f32 window -> bf16 LDS -------------------------
  {
    int t  = tid >> 2;                 // token 0..63
    int c0 = (tid & 3) * 128;
    const float* src = x + winBase + ((long)(t >> 3) * 128 + (t & 7)) * 512 + c0;
    __bf16* dst = Xb + t * 512 + c0;
#pragma unroll
    for (int i = 0; i < 32; ++i) {
      v4f v = *(const v4f*)(src + i * 4);
      dst[i * 4 + 0] = (__bf16)v.x;
      dst[i * 4 + 1] = (__bf16)v.y;
      dst[i * 4 + 2] = (__bf16)v.z;
      dst[i * 4 + 3] = (__bf16)v.w;
    }
  }
  __syncthreads();

  // ---- Stage B: QKV = X @ Wqkv + b  (64x512 @ 512x1536) --------------------
  // M-blocked: one weight B-tile feeds 4 independent accumulators.
  for (int nt = wv; nt < 96; nt += NW) {
    int n0 = nt * 16;
    v8f acc[4] = {};
#pragma unroll 2
    for (int k0 = 0; k0 < 512; k0 += 32) {
      v16bf bm = load_b32x16(WqkvT, 512, n0, k0, lane);
      v16bf a0 = load_a16x32(Xb, 512,  0, k0, lane);
      v16bf a1 = load_a16x32(Xb, 512, 16, k0, lane);
      v16bf a2 = load_a16x32(Xb, 512, 32, k0, lane);
      v16bf a3 = load_a16x32(Xb, 512, 48, k0, lane);
      acc[0] = wmma_bf16(a0, bm, acc[0]);
      acc[1] = wmma_bf16(a1, bm, acc[1]);
      acc[2] = wmma_bf16(a2, bm, acc[2]);
      acc[3] = wmma_bf16(a3, bm, acc[3]);
    }
    int n  = lane & 15;
    int rb = (lane & 16) >> 1;
    int c  = n0 + n;
    float bias = b_qkv[c];
    if (c < 512) {                                  // Q (scaled by 1/sqrt(64))
#pragma unroll
      for (int im = 0; im < 4; ++im)
#pragma unroll
        for (int r = 0; r < 8; ++r)
          Qb[(im * 16 + rb + r) * 512 + c] = (__bf16)((acc[im][r] + bias) * 0.125f);
    } else if (c < 1024) {                          // K
#pragma unroll
      for (int im = 0; im < 4; ++im)
#pragma unroll
        for (int r = 0; r < 8; ++r)
          Kb[(im * 16 + rb + r) * 512 + (c - 512)] = (__bf16)(acc[im][r] + bias);
    } else {                                        // V stored transposed
      int cv = c - 1024, h = cv >> 6, d = cv & 63;
#pragma unroll
      for (int im = 0; im < 4; ++im)
#pragma unroll
        for (int r = 0; r < 8; ++r)
          Vt[h * 4096 + d * 64 + (im * 16 + rb + r)] = (__bf16)(acc[im][r] + bias);
    }
  }
  __syncthreads();

  // ---- Attention: 2 batches of 4 heads (2 waves per head) ------------------
  const int mh = (wv & 1) * 32;                    // this wave's M half (2 tiles)
  for (int bb = 0; bb < 2; ++bb) {
    int h    = bb * 4 + (wv >> 1);
    int slot = wv >> 1;
    float* S = Sc + slot * 4096;

    // scores = Qs @ K^T   (64x64, K-dim = 64); 2x4 register tiles per wave
    {
      v8f acc[2][4] = {};
#pragma unroll
      for (int k0 = 0; k0 < 64; k0 += 32) {
        v16bf a0 = load_a16x32(Qb, 512, mh,      h * 64 + k0, lane);
        v16bf a1 = load_a16x32(Qb, 512, mh + 16, h * 64 + k0, lane);
#pragma unroll
        for (int jn = 0; jn < 4; ++jn) {
          v16bf bm = load_b32x16(Kb, 512, jn * 16, h * 64 + k0, lane);
          acc[0][jn] = wmma_bf16(a0, bm, acc[0][jn]);
          acc[1][jn] = wmma_bf16(a1, bm, acc[1][jn]);
        }
      }
      int n = lane & 15, rb = (lane & 16) >> 1;
#pragma unroll
      for (int im = 0; im < 2; ++im)
#pragma unroll
        for (int jn = 0; jn < 4; ++jn)
#pragma unroll
          for (int r = 0; r < 8; ++r)
            S[(mh + im * 16 + rb + r) * 64 + jn * 16 + n] = acc[im][jn][r];
    }
    __syncthreads();

    // softmax: 256 threads = 4 heads x 64 rows, one row per thread
    {
      float* Srow = Sc + (tid >> 6) * 4096 + (tid & 63) * 64;
      float mx = -1e30f;
#pragma unroll 8
      for (int j = 0; j < 64; ++j) mx = fmaxf(mx, Srow[j]);
      float sum = 0.f;
#pragma unroll 8
      for (int j = 0; j < 64; ++j) { float e = __expf(Srow[j] - mx); Srow[j] = e; sum += e; }
      float inv = 1.f / sum;
#pragma unroll 8
      for (int j = 0; j < 64; ++j) Srow[j] *= inv;
    }
    __syncthreads();

    // O = P @ V -> overwrite Q's LDS slot; A-tile builds amortized over 4 N
    {
      v8f acc[2][4] = {};
#pragma unroll
      for (int k0 = 0; k0 < 64; k0 += 32) {
        v16bf a01[2];
#pragma unroll
        for (int im = 0; im < 2; ++im) {
          int m = lane & 15, kh = (lane & 16) >> 1;
          const float* p = S + (mh + im * 16 + m) * 64 + k0 + kh;
#pragma unroll
          for (int i = 0; i < 8; ++i) {
            a01[im][i] = (__bf16)p[i];
            a01[im][i + 8] = (__bf16)p[i + 16];
          }
        }
#pragma unroll
        for (int dn = 0; dn < 4; ++dn) {
          v16bf bm = load_b32x16(Vt + h * 4096, 64, dn * 16, k0, lane);
          acc[0][dn] = wmma_bf16(a01[0], bm, acc[0][dn]);
          acc[1][dn] = wmma_bf16(a01[1], bm, acc[1][dn]);
        }
      }
      int n = lane & 15, rb = (lane & 16) >> 1;
#pragma unroll
      for (int im = 0; im < 2; ++im)
#pragma unroll
        for (int dn = 0; dn < 4; ++dn)
#pragma unroll
          for (int r = 0; r < 8; ++r)
            Qb[(mh + im * 16 + rb + r) * 512 + h * 64 + dn * 16 + n] =
                (__bf16)acc[im][dn][r];
    }
    __syncthreads();
  }

  // ---- Stage F: out = O @ Wproj + b, scatter back to (B,H,W,C) -------------
  for (int nt = wv; nt < 32; nt += NW) {
    int n0 = nt * 16;
    v8f acc[4] = {};
#pragma unroll 2
    for (int k0 = 0; k0 < 512; k0 += 32) {
      v16bf bm = load_b32x16(WprojT, 512, n0, k0, lane);
      v16bf a0 = load_a16x32(Qb, 512,  0, k0, lane);
      v16bf a1 = load_a16x32(Qb, 512, 16, k0, lane);
      v16bf a2 = load_a16x32(Qb, 512, 32, k0, lane);
      v16bf a3 = load_a16x32(Qb, 512, 48, k0, lane);
      acc[0] = wmma_bf16(a0, bm, acc[0]);
      acc[1] = wmma_bf16(a1, bm, acc[1]);
      acc[2] = wmma_bf16(a2, bm, acc[2]);
      acc[3] = wmma_bf16(a3, bm, acc[3]);
    }
    int n = lane & 15, rb = (lane & 16) >> 1;
    float bias = b_proj[n0 + n];
#pragma unroll
    for (int im = 0; im < 4; ++im)
#pragma unroll
      for (int r = 0; r < 8; ++r) {
        int tkn = im * 16 + rb + r;
        long off = winBase + ((long)(tkn >> 3) * 128 + (tkn & 7)) * 512 + n0 + n;
        out[off] = acc[im][r] + bias;
      }
  }
}

extern "C" void kernel_launch(void* const* d_in, const int* in_sizes, int n_in,
                              void* d_out, int out_size, void* d_ws, size_t ws_size,
                              hipStream_t stream) {
  const float* x     = (const float*)d_in[0];
  const float* Wqkv  = (const float*)d_in[1];
  const float* bqkv  = (const float*)d_in[2];
  const float* Wproj = (const float*)d_in[3];
  const float* bproj = (const float*)d_in[4];
  // d_in[5]/d_in[6] are H=128, W=128 (hard-coded in the kernel grid math).

  __bf16* WqkvT  = (__bf16*)d_ws;                 // 1536*512 bf16 = 1.5 MB
  __bf16* WprojT = WqkvT + 512 * 1536;            //  512*512 bf16 = 0.5 MB
  float* out = (float*)d_out;

  convert_weights<<<(512 * 1536 + 512 * 512 + 255) / 256, 256, 0, stream>>>(
      Wqkv, Wproj, WqkvT, WprojT);

  const size_t lds_bytes = 256 * 1024;            // 4 x 64KB regions (<=320KB/WGP)
  (void)hipFuncSetAttribute((const void*)wmsa_kernel,
                            hipFuncAttributeMaxDynamicSharedMemorySize,
                            (int)lds_bytes);
  wmsa_kernel<<<2048, 256, lds_bytes, stream>>>(x, WqkvT, WprojT, bqkv, bproj, out);
  (void)in_sizes; (void)n_in; (void)out_size; (void)ws_size;
}
